// RGCNEncoder_3066606649991
// MI455X (gfx1250) — matrix operand order repaired and checked by reading, not compiled, not run on hardware
//
#include <hip/hip_runtime.h>
#include <hip/hip_bf16.h>
#include <math.h>

// ---------------------------------------------------------------------------
// RGCN encoder for MI455X (gfx1250, wave32, WMMA).
// Dense GEMMs: V_WMMA_F32_16X16X32_BF16 with bf16 hi/lo split (bf16x3) for
// near-fp32 accuracy. A/B operands are pre-split/pre-transposed to bf16 planes
// so the GEMM hot loop is pure: async global->LDS copies (ASYNCcnt) feeding
// ds_load_b128 fragment loads into WMMA, double-buffered.
// Sparse aggregation: transform-then-gather + HW global_atomic_add_f32.
// ---------------------------------------------------------------------------

typedef __attribute__((ext_vector_type(16))) __bf16 v16bf;
typedef __attribute__((ext_vector_type(8)))  __bf16 v8bf;
typedef __attribute__((ext_vector_type(8)))  float  v8f;

#define N_REL 5
#define N_BASES 4

#define BM 64
#define BN 128
#define BK 32
#define APAD 8
#define BPAD 8

// ------------------------- low-level helpers -------------------------------

__device__ __forceinline__ v8f wmma_bf16(v16bf a, v16bf b, v8f c) {
    return __builtin_amdgcn_wmma_f32_16x16x32_bf16(
        false, a, false, b, (short)0, c, false, false);
}

__device__ __forceinline__ v16bf ld_frag(const __bf16* p0, const __bf16* p1) {
    v8bf a = *(const v8bf*)p0;   // ds_load_b128
    v8bf b = *(const v8bf*)p1;   // ds_load_b128
    return __builtin_shufflevector(a, b, 0,1,2,3,4,5,6,7,8,9,10,11,12,13,14,15);
}

__device__ __forceinline__ void split_bf16(float x, __bf16& hi, __bf16& lo) {
    hi = (__bf16)x;
    lo = (__bf16)(x - (float)hi);
}

// LDS byte offset of a generic pointer to __shared__ (ISA 10.2: addr[31:0]).
__device__ __forceinline__ unsigned lds_addr(const void* p) {
    return (unsigned)(reinterpret_cast<unsigned long long>(p) & 0xffffffffull);
}

// Async 16B global -> LDS copy (per lane). Tracked by ASYNCcnt.
__device__ __forceinline__ void async_b128(unsigned lds, const __bf16* g) {
    asm volatile("global_load_async_to_lds_b128 %0, %1, off"
                 :: "v"(lds), "v"(g) : "memory");
}

__device__ __forceinline__ void wait_async0() {
    asm volatile("s_wait_asynccnt 0x0" ::: "memory");
}

// ------------------------------ GEMM ---------------------------------------
// Out[M,N] = A[M,K] @ B[K,N] (+bias). A given as bf16 hi/lo planes row-major
// [M,K]; B given pre-transposed bf16 hi/lo planes [N,K] (K contiguous).
// Block: 64x128, BK=32, 256 threads = 8 waves (2x4), wave patch 32x32.

__global__ __launch_bounds__(256)
void rgcn_gemm_bf16x3(const __bf16* __restrict__ Ahi, const __bf16* __restrict__ Alo,
                      const __bf16* __restrict__ BThi, const __bf16* __restrict__ BTlo,
                      const float* __restrict__ bias, float* __restrict__ Out,
                      int M, int K, int N) {
    // [buf][plane: 0=hi 1=lo]
    __shared__ __align__(16) __bf16 As[2][2][BM][BK + APAD];
    __shared__ __align__(16) __bf16 Bs[2][2][BN][BK + BPAD];

    const int row0 = blockIdx.y * BM;
    const int col0 = blockIdx.x * BN;

    const int tid  = threadIdx.x;
    const int lane = tid & 31;
    const int wave = tid >> 5;
    const int wm   = wave >> 2;     // 0..1
    const int wn   = wave & 3;      // 0..3
    const int l16  = lane & 15;
    const int lhi  = lane >> 4;

    const __bf16* Ap[2] = { Ahi, Alo };
    const __bf16* Bp[2] = { BThi, BTlo };

    // per-thread copy chunks: A = 1 x 16B per plane, B = 2 x 16B per plane
    const int ar = tid >> 2, aq = tid & 3;                    // 64 rows x 4 chunks
    const int agr = (row0 + ar < M) ? (row0 + ar) : (M - 1);  // clamp (rows padded)

    auto stage = [&](int buf, int k0) {
#pragma unroll
        for (int p = 0; p < 2; ++p)
            async_b128(lds_addr(&As[buf][p][ar][aq * 8]),
                       Ap[p] + (size_t)agr * (size_t)K + k0 + aq * 8);
#pragma unroll
        for (int s = 0; s < 2; ++s) {
            int c  = tid + s * 256;                            // 128 rows x 4 chunks
            int bn = c >> 2, bq = c & 3;
#pragma unroll
            for (int p = 0; p < 2; ++p)
                async_b128(lds_addr(&Bs[buf][p][bn][bq * 8]),
                           Bp[p] + (size_t)(col0 + bn) * (size_t)K + k0 + bq * 8);
        }
    };

    v8f acc[2][2] = {};
    const int steps = K / BK;

    stage(0, 0);
    for (int i = 0; i < steps; ++i) {
        wait_async0();        // stage(i) landed in LDS
        __syncthreads();      // all waves done with compute(i-1) and see stage(i)
        if (i + 1 < steps) stage((i + 1) & 1, (i + 1) * BK);  // overlap with compute(i)

        const int buf = i & 1;

        // A 16x32 bf16 frag: lanes0-15 K {0..7,16..23}, lanes16-31 K {8..15,24..31}
        v16bf ah[2], al[2];
#pragma unroll
        for (int t = 0; t < 2; ++t) {
            int r_ = wm * 32 + t * 16 + l16;
            ah[t] = ld_frag(&As[buf][0][r_][lhi * 8], &As[buf][0][r_][16 + lhi * 8]);
            al[t] = ld_frag(&As[buf][1][r_][lhi * 8], &As[buf][1][r_][16 + lhi * 8]);
        }
        // B 32x16 bf16 frag: lane = column n, lanes0-15 K {0..7,16..23}, lanes16-31 K {8..15,24..31}
        v16bf bh[2], bl[2];
#pragma unroll
        for (int t = 0; t < 2; ++t) {
            int n_ = wn * 32 + t * 16 + l16;
            bh[t] = ld_frag(&Bs[buf][0][n_][lhi * 8], &Bs[buf][0][n_][16 + lhi * 8]);
            bl[t] = ld_frag(&Bs[buf][1][n_][lhi * 8], &Bs[buf][1][n_][16 + lhi * 8]);
        }

        // bf16x3: hi*hi + hi*lo + lo*hi
#pragma unroll
        for (int ti = 0; ti < 2; ++ti)
#pragma unroll
            for (int tj = 0; tj < 2; ++tj) {
                acc[ti][tj] = wmma_bf16(ah[ti], bh[tj], acc[ti][tj]);
                acc[ti][tj] = wmma_bf16(ah[ti], bl[tj], acc[ti][tj]);
                acc[ti][tj] = wmma_bf16(al[ti], bh[tj], acc[ti][tj]);
            }
    }

    // C/D layout: VGPR v -> M = base + v (lanes0-15), base+8+v handled by lhi
#pragma unroll
    for (int ti = 0; ti < 2; ++ti)
#pragma unroll
        for (int tj = 0; tj < 2; ++tj) {
            int grb = row0 + wm * 32 + ti * 16 + lhi * 8;
            int gc  = col0 + wn * 32 + tj * 16 + l16;
            float b = bias ? bias[gc] : 0.f;
#pragma unroll
            for (int v = 0; v < 8; ++v) {
                int gr = grb + v;
                if (gr < M) Out[(size_t)gr * (size_t)N + gc] = acc[ti][tj][v] + b;
            }
        }
}

// ------------------ operand preparation kernels ----------------------------

// f32 [n] -> bf16 hi/lo planes (row-major preserved)
__global__ void rgcn_split_f32(const float* __restrict__ in, size_t n,
                               __bf16* __restrict__ hi, __bf16* __restrict__ lo) {
    for (size_t i = (size_t)blockIdx.x * blockDim.x + threadIdx.x;
         i < n; i += (size_t)gridDim.x * blockDim.x) {
        __bf16 h, l; split_bf16(in[i], h, l);
        hi[i] = h; lo[i] = l;
    }
}

// W_r = sum_b comp[r,b]*basis[b]  (or identity if comp==nullptr), emitted
// TRANSPOSED as bf16 hi/lo planes: T[n*K + k] for W[k*N + n].
__global__ void rgcn_combine_wT(const float* __restrict__ Wsrc,
                                const float* __restrict__ comp, int r,
                                int K, int N,
                                __bf16* __restrict__ Thi, __bf16* __restrict__ Tlo) {
    const size_t sz = (size_t)K * (size_t)N;
    float c0 = 0.f, c1 = 0.f, c2 = 0.f, c3 = 0.f;
    if (comp) {
        c0 = comp[r * N_BASES + 0]; c1 = comp[r * N_BASES + 1];
        c2 = comp[r * N_BASES + 2]; c3 = comp[r * N_BASES + 3];
    }
    for (size_t i = (size_t)blockIdx.x * blockDim.x + threadIdx.x;
         i < sz; i += (size_t)gridDim.x * blockDim.x) {
        float v;
        if (comp) {
            v = c0 * Wsrc[i];
            v = fmaf(c1, Wsrc[sz + i], v);
            v = fmaf(c2, Wsrc[2 * sz + i], v);
            v = fmaf(c3, Wsrc[3 * sz + i], v);
        } else {
            v = Wsrc[i];
        }
        int k = (int)(i / N), n = (int)(i % N);
        __bf16 h, l; split_bf16(v, h, l);
        size_t o = (size_t)n * (size_t)K + k;
        Thi[o] = h; Tlo[o] = l;
    }
}

// --------------------------- edge counting ---------------------------------
__global__ void rgcn_count_edges(const int* __restrict__ dst,
                                 const int* __restrict__ etype,
                                 int E, float* __restrict__ cnt) {
    int e = blockIdx.x * blockDim.x + threadIdx.x;
    if (e < E) unsafeAtomicAdd(&cnt[(size_t)dst[e] * N_REL + etype[e]], 1.0f);
}

__global__ void rgcn_inv_counts(float* __restrict__ cnt, int n) {
    int i = blockIdx.x * blockDim.x + threadIdx.x;
    if (i < n) cnt[i] = 1.0f / fmaxf(cnt[i], 1.0f);
}

// ------------------------- scatter (mean agg) ------------------------------
__global__ void rgcn_scatter_rel(const float* __restrict__ H,
                                 const int* __restrict__ src,
                                 const int* __restrict__ dst,
                                 const int* __restrict__ etype,
                                 const float* __restrict__ invcnt,
                                 int F, int r, float* __restrict__ Out) {
    int e = blockIdx.x;
    if (etype[e] != r) return;
    int s = src[e], d = dst[e];
    float scale = invcnt[(size_t)d * N_REL + r];
    const float4* hp = (const float4*)(H + (size_t)s * F);
    float* op = Out + (size_t)d * F;
    for (int q = threadIdx.x; q < (F >> 2); q += blockDim.x) {
        float4 v = hp[q];
        unsafeAtomicAdd(op + q * 4 + 0, v.x * scale);  // global_atomic_add_f32
        unsafeAtomicAdd(op + q * 4 + 1, v.y * scale);
        unsafeAtomicAdd(op + q * 4 + 2, v.z * scale);
        unsafeAtomicAdd(op + q * 4 + 3, v.w * scale);
    }
}

// --------------- BN(eval) + ReLU fused with bf16 hi/lo split ---------------
__global__ void rgcn_bn_relu_split(const float* __restrict__ in,
                                   const float* __restrict__ gamma,
                                   const float* __restrict__ beta,
                                   const float* __restrict__ mean,
                                   const float* __restrict__ var,
                                   __bf16* __restrict__ hhi, __bf16* __restrict__ hlo,
                                   size_t total, int F) {
    for (size_t i = (size_t)blockIdx.x * blockDim.x + threadIdx.x;
         i < total; i += (size_t)gridDim.x * blockDim.x) {
        int c = (int)(i % F);
        float v = (in[i] - mean[c]) * rsqrtf(var[c] + 1e-5f) * gamma[c] + beta[c];
        v = fmaxf(v, 0.f);
        __bf16 h, l; split_bf16(v, h, l);
        hhi[i] = h; hlo[i] = l;
    }
}

// --------------------------- L2 normalize ----------------------------------
__global__ void rgcn_l2norm(const float* __restrict__ in,
                            float* __restrict__ out, int F) {
    const int row = blockIdx.x;
    const float* p = in + (size_t)row * F;
    float s = 0.f;
    for (int q = threadIdx.x; q < F; q += blockDim.x) { float v = p[q]; s += v * v; }
    for (int off = 16; off > 0; off >>= 1) s += __shfl_xor(s, off, 32);
    __shared__ float red[8];
    int wv = threadIdx.x >> 5, ln = threadIdx.x & 31;
    if (ln == 0) red[wv] = s;
    __syncthreads();
    if (threadIdx.x == 0) {
        float t = 0.f;
        for (int i = 0; i < (int)(blockDim.x >> 5); ++i) t += red[i];
        red[0] = t;
    }
    __syncthreads();
    float inv = 1.f / fmaxf(sqrtf(red[0]), 1e-12f);
    float* o = out + (size_t)row * F;
    for (int q = threadIdx.x; q < F; q += blockDim.x) o[q] = p[q] * inv;
}

// ------------------------------ launcher -----------------------------------
static inline size_t align_up(size_t x, size_t a) { return (x + a - 1) & ~(a - 1); }

extern "C" void kernel_launch(void* const* d_in, const int* in_sizes, int n_in,
                              void* d_out, int out_size, void* d_ws, size_t ws_size,
                              hipStream_t stream) {
    const float* x      = (const float*)d_in[0];
    const int*   eidx   = (const int*)d_in[1];
    const int*   etype  = (const int*)d_in[2];
    const float* basis1 = (const float*)d_in[3];
    const float* comp1  = (const float*)d_in[4];
    const float* root1  = (const float*)d_in[5];
    const float* bias1  = (const float*)d_in[6];
    const float* gamma  = (const float*)d_in[7];
    const float* beta   = (const float*)d_in[8];
    const float* mean   = (const float*)d_in[9];
    const float* var    = (const float*)d_in[10];
    const float* basis2 = (const float*)d_in[11];
    const float* comp2  = (const float*)d_in[12];
    const float* root2  = (const float*)d_in[13];
    const float* bias2  = (const float*)d_in[14];

    const int IN_CH = 256, HID = 768, OUT = 768;
    const int Nn = in_sizes[0] / IN_CH;       // 50000
    const int E  = in_sizes[2];               // 200000
    const int* src = eidx;                    // edge_index[0,:]
    const int* dst = eidx + E;                // edge_index[1,:]

    // workspace layout
    char* w = (char*)d_ws;
    float*  bufA = (float*)w;  w += align_up((size_t)Nn * HID * 4, 256);   // accumulator
    float*  bufB = (float*)w;  w += align_up((size_t)Nn * HID * 4, 256);   // H_r transform
    __bf16* Xhi  = (__bf16*)w; w += align_up((size_t)Nn * IN_CH * 2, 256);
    __bf16* Xlo  = (__bf16*)w; w += align_up((size_t)Nn * IN_CH * 2, 256);
    __bf16* Hhi  = (__bf16*)w; w += align_up((size_t)Nn * HID * 2, 256);
    __bf16* Hlo  = (__bf16*)w; w += align_up((size_t)Nn * HID * 2, 256);
    __bf16* WThi = (__bf16*)w; w += align_up((size_t)HID * OUT * 2, 256);
    __bf16* WTlo = (__bf16*)w; w += align_up((size_t)HID * OUT * 2, 256);
    float*  cnt  = (float*)w;  w += align_up((size_t)Nn * N_REL * 4, 256);

    // per-(node, relation) in-degree -> inverse
    hipMemsetAsync(cnt, 0, (size_t)Nn * N_REL * 4, stream);
    rgcn_count_edges<<<(E + 255) / 256, 256, 0, stream>>>(dst, etype, E, cnt);
    rgcn_inv_counts<<<(Nn * N_REL + 255) / 256, 256, 0, stream>>>(cnt, Nn * N_REL);

    // pre-split X once (reused by 6 layer-1 GEMMs)
    rgcn_split_f32<<<2048, 256, 0, stream>>>(x, (size_t)Nn * IN_CH, Xhi, Xlo);

    const dim3 gemm_grid((HID + BN - 1) / BN, (Nn + BM - 1) / BM);

    // ---------------- layer 1 ----------------
    rgcn_combine_wT<<<1024, 256, 0, stream>>>(root1, nullptr, 0, IN_CH, HID, WThi, WTlo);
    rgcn_gemm_bf16x3<<<gemm_grid, 256, 0, stream>>>(Xhi, Xlo, WThi, WTlo, bias1, bufA,
                                                    Nn, IN_CH, HID);
    for (int r = 0; r < N_REL; ++r) {
        rgcn_combine_wT<<<1024, 256, 0, stream>>>(basis1, comp1, r, IN_CH, HID, WThi, WTlo);
        rgcn_gemm_bf16x3<<<gemm_grid, 256, 0, stream>>>(Xhi, Xlo, WThi, WTlo, nullptr, bufB,
                                                        Nn, IN_CH, HID);
        rgcn_scatter_rel<<<E, 192, 0, stream>>>(bufB, src, dst, etype, cnt, HID, r, bufA);
    }
    rgcn_bn_relu_split<<<2048, 256, 0, stream>>>(bufA, gamma, beta, mean, var,
                                                 Hhi, Hlo, (size_t)Nn * HID, HID);

    // ---------------- layer 2 ----------------
    rgcn_combine_wT<<<1024, 256, 0, stream>>>(root2, nullptr, 0, HID, OUT, WThi, WTlo);
    rgcn_gemm_bf16x3<<<gemm_grid, 256, 0, stream>>>(Hhi, Hlo, WThi, WTlo, bias2, bufA,
                                                    Nn, HID, OUT);
    for (int r = 0; r < N_REL; ++r) {
        rgcn_combine_wT<<<1024, 256, 0, stream>>>(basis2, comp2, r, HID, OUT, WThi, WTlo);
        rgcn_gemm_bf16x3<<<gemm_grid, 256, 0, stream>>>(Hhi, Hlo, WThi, WTlo, nullptr, bufB,
                                                        Nn, HID, OUT);
        rgcn_scatter_rel<<<E, 192, 0, stream>>>(bufB, src, dst, etype, cnt, OUT, r, bufA);
    }
    rgcn_l2norm<<<Nn, 256, 0, stream>>>(bufA, (float*)d_out, OUT);
}
